// TriangularUpdate_54211077210610
// MI455X (gfx1250) — matrix operand order repaired and checked by reading, not compiled, not run on hardware
//
#include <hip/hip_runtime.h>
#include <cstdint>
#include <cstddef>

typedef __attribute__((ext_vector_type(16))) __bf16 v16bf;
typedef __attribute__((ext_vector_type(8)))  __bf16 bf16x8;
typedef __attribute__((ext_vector_type(4)))  __bf16 bf16x4;
typedef __attribute__((ext_vector_type(8)))  float  v8f;

#define NN   512
#define EE   128
#define CC   128
#define PP   (512*512)      // 262144 flattened (i,j) positions

__device__ __forceinline__ float sigmoidf_(float x) {
    return 1.0f / (1.0f + __expf(-x));
}

// ---- WMMA fragment loaders (CDNA5 ISA 7.12.2 layouts) -----------------------
// A: 16x32 bf16, row-major source with leading dim lda (elements).
// lane L: row m = L%16 ; holds K=[h*8, h*8+8) and K=[16+h*8, 16+h*8+8), h=L/16.
__device__ __forceinline__ v16bf load_a_frag(const __bf16* base, int lda, int lane) {
    int m = lane & 15;
    int h = (lane >> 4) << 3;                 // 0 or 8
    const __bf16* r = base + (size_t)m * lda + h;
    union { v16bf v; bf16x8 q[2]; } u;
    u.q[0] = *(const bf16x8*)(r);             // K = h .. h+7
    u.q[1] = *(const bf16x8*)(r + 16);        // K = 16+h .. 16+h+7
    return u.v;
}

// B: 32x16 bf16, B[k][n] = W[n][k]; W row-major over k with leading dim ldb.
// lane L: column n = L%16 ; holds K=[h*8, h*8+8) and K=[16+h*8, 16+h*8+8), h=L/16.
__device__ __forceinline__ v16bf load_b_frag(const __bf16* base, int ldb, int lane) {
    int n = lane & 15;
    int h = (lane >> 4) << 3;                 // 0 or 8
    const __bf16* r = base + (size_t)n * ldb + h;
    union { v16bf v; bf16x8 q[2]; } u;
    u.q[0] = *(const bf16x8*)(r);
    u.q[1] = *(const bf16x8*)(r + 16);
    return u.v;
}

// ---- Kernel 0: f32 -> bf16 weight conversion (6 x [128,128]) ----------------
__global__ __launch_bounds__(256) void k_convert(
    const float* __restrict__ a, const float* __restrict__ b,
    const float* __restrict__ c, const float* __restrict__ d,
    const float* __restrict__ e, const float* __restrict__ f,
    __bf16* __restrict__ out)
{
    int i = blockIdx.x * 256 + threadIdx.x;       // 0 .. 98303
    int m = i >> 14, k = i & 16383;
    float v;
    if      (m == 0) v = a[k];
    else if (m == 1) v = b[k];
    else if (m == 2) v = c[k];
    else if (m == 3) v = d[k];
    else if (m == 4) v = e[k];
    else             v = f[k];
    out[i] = (__bf16)v;
}

// ---- Kernel 1: LayerNorm(E) + 5 projections + gating ------------------------
// One wave per 16-position tile. Writes lu/ru channel-major bf16, fu [p][e] f32.
__global__ __launch_bounds__(128) void k_proj(
    const float* __restrict__ x, const float* __restrict__ mask,
    const __bf16* __restrict__ wle, const __bf16* __restrict__ wre,
    const __bf16* __restrict__ wlu, const __bf16* __restrict__ wru,
    const __bf16* __restrict__ wfu,
    const float* __restrict__ ble, const float* __restrict__ bre,
    const float* __restrict__ blu, const float* __restrict__ bru,
    const float* __restrict__ bfu,
    const float* __restrict__ eng, const float* __restrict__ enb,
    __bf16* __restrict__ lut, __bf16* __restrict__ rut,
    float* __restrict__ fuo)
{
    __shared__ __bf16 xt[4][16][136];
    __shared__ float  smask[4][16];

    const int lane = threadIdx.x & 31;
    const int wv   = threadIdx.x >> 5;
    const int p0   = (blockIdx.x * 4 + wv) * 16;
    const int row  = lane & 15;
    const int half = lane >> 4;

    // ---- LayerNorm: 2 lanes per row, 64 elements each ----
    const float4* rp = (const float4*)(x + (size_t)(p0 + row) * EE) + half * 16;
    float s = 0.f, s2 = 0.f;
#pragma unroll
    for (int q = 0; q < 16; ++q) {
        float4 v = rp[q];
        s  += v.x + v.y + v.z + v.w;
        s2 += v.x*v.x + v.y*v.y + v.z*v.z + v.w*v.w;
    }
    s  += __shfl_xor(s, 16, 32);
    s2 += __shfl_xor(s2, 16, 32);
    float mean = s * (1.f / 128.f);
    float inv  = rsqrtf(s2 * (1.f / 128.f) - mean * mean + 1e-5f);

    const float4* gp = (const float4*)eng + half * 16;
    const float4* bp = (const float4*)enb + half * 16;
#pragma unroll
    for (int q = 0; q < 16; ++q) {
        float4 v = rp[q]; float4 g = gp[q]; float4 b = bp[q];
        bf16x4 o;
        o[0] = (__bf16)((v.x - mean) * inv * g.x + b.x);
        o[1] = (__bf16)((v.y - mean) * inv * g.y + b.y);
        o[2] = (__bf16)((v.z - mean) * inv * g.z + b.z);
        o[3] = (__bf16)((v.w - mean) * inv * g.w + b.w);
        *(bf16x4*)&xt[wv][row][half * 64 + q * 4] = o;
    }
    if (lane < 16) smask[wv][lane] = mask[p0 + lane];
    __syncthreads();

    v16bf afr[4];
#pragma unroll
    for (int kc = 0; kc < 4; ++kc)
        afr[kc] = load_a_frag(&xt[wv][0][kc * 32], 136, lane);

    const int cn = lane & 15;
    const int hh = lane >> 4;

#pragma unroll
    for (int nt = 0; nt < 8; ++nt) {
        const int n0 = nt * 16;
        v8f aLE, aRE, aLU, aRU, aFU;
        float vle = ble[n0 + cn], vre = bre[n0 + cn], vlu = blu[n0 + cn];
        float vru = bru[n0 + cn], vfu = bfu[n0 + cn];
#pragma unroll
        for (int v = 0; v < 8; ++v) { aLE[v]=vle; aRE[v]=vre; aLU[v]=vlu; aRU[v]=vru; aFU[v]=vfu; }

#pragma unroll
        for (int kc = 0; kc < 4; ++kc) {
            v16bf bLE = load_b_frag(wle + (size_t)n0 * EE + kc * 32, EE, lane);
            v16bf bRE = load_b_frag(wre + (size_t)n0 * EE + kc * 32, EE, lane);
            v16bf bLU = load_b_frag(wlu + (size_t)n0 * EE + kc * 32, EE, lane);
            v16bf bRU = load_b_frag(wru + (size_t)n0 * EE + kc * 32, EE, lane);
            v16bf bFU = load_b_frag(wfu + (size_t)n0 * EE + kc * 32, EE, lane);
            aLE = __builtin_amdgcn_wmma_f32_16x16x32_bf16(false, afr[kc], false, bLE, (short)0, aLE, false, false);
            aRE = __builtin_amdgcn_wmma_f32_16x16x32_bf16(false, afr[kc], false, bRE, (short)0, aRE, false, false);
            aLU = __builtin_amdgcn_wmma_f32_16x16x32_bf16(false, afr[kc], false, bLU, (short)0, aLU, false, false);
            aRU = __builtin_amdgcn_wmma_f32_16x16x32_bf16(false, afr[kc], false, bRU, (short)0, aRU, false, false);
            aFU = __builtin_amdgcn_wmma_f32_16x16x32_bf16(false, afr[kc], false, bFU, (short)0, aFU, false, false);
        }

        // gating epilogue; channel-major stores for lu/ru
        bf16x8 luv, ruv;
#pragma unroll
        for (int v = 0; v < 8; ++v) {
            int m = hh * 8 + v;
            float mk = smask[wv][m];
            float le = aLE[v] * mk;
            float re = aRE[v] * mk;
            luv[v] = (__bf16)(le * sigmoidf_(aLU[v]));
            ruv[v] = (__bf16)(sigmoidf_(aRU[v]) * re);
            fuo[(size_t)(p0 + m) * CC + n0 + cn] = sigmoidf_(aFU[v]);
        }
        *(bf16x8*)(lut + (size_t)(n0 + cn) * PP + p0 + hh * 8) = luv;
        *(bf16x8*)(rut + (size_t)(n0 + cn) * PP + p0 + hh * 8) = ruv;
    }
}

// ---- Kernel 2: per-channel triangle GEMM  U_c = Lu_c * Ru_c^T ---------------
// grid = (32, 128): x -> (i_blk 0..3, j_blk 0..7), y -> channel.
// block tile 128x64, wave tile 32x32 (2x2 WMMA), K loop 512/32.
__global__ __launch_bounds__(256) void k_tri(
    const __bf16* __restrict__ lut, const __bf16* __restrict__ rut,
    float* __restrict__ u)
{
    const int lane = threadIdx.x & 31;
    const int wv   = threadIdx.x >> 5;
    const int c    = blockIdx.y;
    const int i0   = (blockIdx.x >> 3) * 128 + (wv >> 1) * 32;
    const int j0   = (blockIdx.x & 7)  * 64  + (wv & 1)  * 32;

    const __bf16* A = lut + (size_t)c * PP;
    const __bf16* B = rut + (size_t)c * PP;

    v8f acc[2][2];
#pragma unroll
    for (int ti = 0; ti < 2; ++ti)
#pragma unroll
        for (int tj = 0; tj < 2; ++tj)
#pragma unroll
            for (int v = 0; v < 8; ++v) acc[ti][tj][v] = 0.f;

    for (int k = 0; k < NN; k += 32) {
        v16bf af0 = load_a_frag(A + (size_t)(i0)      * NN + k, NN, lane);
        v16bf af1 = load_a_frag(A + (size_t)(i0 + 16) * NN + k, NN, lane);
        v16bf bf0 = load_b_frag(B + (size_t)(j0)      * NN + k, NN, lane);
        v16bf bf1 = load_b_frag(B + (size_t)(j0 + 16) * NN + k, NN, lane);
        acc[0][0] = __builtin_amdgcn_wmma_f32_16x16x32_bf16(false, af0, false, bf0, (short)0, acc[0][0], false, false);
        acc[0][1] = __builtin_amdgcn_wmma_f32_16x16x32_bf16(false, af0, false, bf1, (short)0, acc[0][1], false, false);
        acc[1][0] = __builtin_amdgcn_wmma_f32_16x16x32_bf16(false, af1, false, bf0, (short)0, acc[1][0], false, false);
        acc[1][1] = __builtin_amdgcn_wmma_f32_16x16x32_bf16(false, af1, false, bf1, (short)0, acc[1][1], false, false);
    }

    const int n = lane & 15, h = lane >> 4;
    float* U = u + (size_t)c * PP;
#pragma unroll
    for (int ti = 0; ti < 2; ++ti)
#pragma unroll
        for (int tj = 0; tj < 2; ++tj)
#pragma unroll
            for (int v = 0; v < 8; ++v)
                U[(size_t)(i0 + ti * 16 + h * 8 + v) * NN + j0 + tj * 16 + n] = acc[ti][tj][v];
}

// ---- Kernel 3: LayerNorm(C) + rs projection + *fu ---------------------------
__global__ __launch_bounds__(128) void k_fin(
    const float* __restrict__ u, const float* __restrict__ fu,
    const __bf16* __restrict__ wrs, const float* __restrict__ brs,
    const float* __restrict__ cng, const float* __restrict__ cnb,
    float* __restrict__ out)
{
    __shared__ float  uf[4][16][132];
    __shared__ __bf16 xb[4][16][136];

    const int lane = threadIdx.x & 31;
    const int wv   = threadIdx.x >> 5;
    const int p0   = (blockIdx.x * 4 + wv) * 16;
    const int row  = lane & 15;
    const int half = lane >> 4;

    // gather transpose: channel-major u -> [p][c] LDS tile (coalesced 64B runs)
#pragma unroll 8
    for (int it = 0; it < 64; ++it) {
        int c = it * 2 + half;
        uf[wv][row][c] = u[(size_t)c * PP + p0 + row];
    }
    __syncthreads();

    float s = 0.f, s2 = 0.f;
#pragma unroll
    for (int q = 0; q < 64; ++q) {
        float v = uf[wv][row][half * 64 + q];
        s += v; s2 += v * v;
    }
    s  += __shfl_xor(s, 16, 32);
    s2 += __shfl_xor(s2, 16, 32);
    float mean = s * (1.f / 128.f);
    float inv  = rsqrtf(s2 * (1.f / 128.f) - mean * mean + 1e-5f);
#pragma unroll
    for (int q = 0; q < 64; ++q) {
        int cix = half * 64 + q;
        float v = uf[wv][row][cix];
        xb[wv][row][cix] = (__bf16)((v - mean) * inv * cng[cix] + cnb[cix]);
    }
    __syncthreads();

    v16bf afr[4];
#pragma unroll
    for (int kc = 0; kc < 4; ++kc)
        afr[kc] = load_a_frag(&xb[wv][0][kc * 32], 136, lane);

    const int cn = lane & 15;
    const int hh = lane >> 4;
#pragma unroll
    for (int nt = 0; nt < 8; ++nt) {
        const int n0 = nt * 16;
        v8f acc;
        float bb = brs[n0 + cn];
#pragma unroll
        for (int v = 0; v < 8; ++v) acc[v] = bb;
#pragma unroll
        for (int kc = 0; kc < 4; ++kc) {
            v16bf b = load_b_frag(wrs + (size_t)n0 * CC + kc * 32, CC, lane);
            acc = __builtin_amdgcn_wmma_f32_16x16x32_bf16(false, afr[kc], false, b, (short)0, acc, false, false);
        }
#pragma unroll
        for (int v = 0; v < 8; ++v) {
            int p = p0 + hh * 8 + v;
            int e = n0 + cn;
            out[(size_t)p * EE + e] = acc[v] * fu[(size_t)p * EE + e];
        }
    }
}

// ---- launch -----------------------------------------------------------------
extern "C" void kernel_launch(void* const* d_in, const int* in_sizes, int n_in,
                              void* d_out, int out_size, void* d_ws, size_t ws_size,
                              hipStream_t stream) {
    const float* pair = (const float*)d_in[0];
    const float* mask = (const float*)d_in[1];
    const float* le_w = (const float*)d_in[2];  const float* le_b = (const float*)d_in[3];
    const float* re_w = (const float*)d_in[4];  const float* re_b = (const float*)d_in[5];
    const float* lu_w = (const float*)d_in[6];  const float* lu_b = (const float*)d_in[7];
    const float* ru_w = (const float*)d_in[8];  const float* ru_b = (const float*)d_in[9];
    const float* fu_w = (const float*)d_in[10]; const float* fu_b = (const float*)d_in[11];
    const float* rs_w = (const float*)d_in[12]; const float* rs_b = (const float*)d_in[13];
    const float* en_g = (const float*)d_in[14]; const float* en_b = (const float*)d_in[15];
    const float* cn_g = (const float*)d_in[16]; const float* cn_b = (const float*)d_in[17];
    float* outp = (float*)d_out;

    char* ws = (char*)d_ws;
    // ws layout (bytes):
    //   [0)        6 bf16 weight matrices, 6*16384*2 = 196608
    //   lu_t bf16  channel-major [128][262144]  : 67108864
    //   ru_t bf16  channel-major                : 67108864
    //   fu   f32   [262144][128]                : 134217728
    //   u    f32   channel-major [128][262144]  : 134217728
    __bf16* wbf  = (__bf16*)ws;
    __bf16* wle  = wbf;
    __bf16* wre  = wbf + 16384;
    __bf16* wlu  = wbf + 2 * 16384;
    __bf16* wru  = wbf + 3 * 16384;
    __bf16* wfu  = wbf + 4 * 16384;
    __bf16* wrs  = wbf + 5 * 16384;
    __bf16* lut  = (__bf16*)(ws + 196608);
    __bf16* rut  = (__bf16*)(ws + 196608 + 67108864ull);
    float*  fub  = (float*)(ws + 196608 + 134217728ull);
    float*  ubuf = (float*)(ws + 196608 + 134217728ull + 134217728ull);

    k_convert<<<384, 256, 0, stream>>>(le_w, re_w, lu_w, ru_w, fu_w, rs_w, wbf);

    k_proj<<<4096, 128, 0, stream>>>(pair, mask, wle, wre, wlu, wru, wfu,
                                     le_b, re_b, lu_b, ru_b, fu_b,
                                     en_g, en_b, lut, rut, fub);

    k_tri<<<dim3(32, 128), 256, 0, stream>>>(lut, rut, ubuf);

    k_fin<<<4096, 128, 0, stream>>>(ubuf, fub, wrs, rs_b, cn_g, cn_b, outp);
}
